// TranslationTransformer_24893630447832
// MI455X (gfx1250) — matrix operand order repaired and checked by reading, not compiled, not run on hardware
//
#include <hip/hip_runtime.h>

// ---------------------------------------------------------------------------
// Model dims (fixed by the reference)
// ---------------------------------------------------------------------------
constexpr int L = 6, H = 8, C = 512, D = 64, F = 2048, V = 32000, B = 2, T = 1024;
constexpr int M    = B * T;       // 2048 token rows
constexpr int NQKV = 3 * H * D;   // 1536 fused qkv columns

typedef __attribute__((ext_vector_type(16))) _Float16 v16h;
typedef __attribute__((ext_vector_type(8)))  float    v8f;
typedef __attribute__((ext_vector_type(4)))  float    f4;
typedef __attribute__((ext_vector_type(4)))  _Float16 h4;
typedef __attribute__((ext_vector_type(4)))  unsigned int v4u;
typedef __attribute__((ext_vector_type(8)))  int      v8i;
typedef __attribute__((ext_vector_type(4)))  int      v4i;

// ---------------------------------------------------------------------------
// WMMA helpers: CDNA5 16x16x32 f16 -> f32  (CDNA5 ISA layouts)
// ---------------------------------------------------------------------------
__device__ __forceinline__ v8f wmma16(v16h a, v16h b, v8f c) {
  return __builtin_amdgcn_wmma_f32_16x16x32_f16(false, a, false, b, (short)0, c,
                                                false, false);
}

__device__ __forceinline__ h4 cvt4(f4 v) {
  h4 r;
  r[0] = (_Float16)v[0]; r[1] = (_Float16)v[1];
  r[2] = (_Float16)v[2]; r[3] = (_Float16)v[3];
  return r;
}

// A-matrix 16x32 (M x K) f16 fragment from row-major LDS tile (ld = halfs/row).
__device__ __forceinline__ v16h frag_a(const _Float16* base, int ld, int lane) {
  const int m  = lane & 15;
  const int hb = (lane >> 4) << 4;
  v16h a;
#pragma unroll
  for (int e = 0; e < 16; ++e) {
    const int k = e + hb;
    a[e] = base[m * ld + k];
  }
  return a;
}

// B-matrix 32x16 (K x N) fragment, B stored row-major [K][N] in LDS.
__device__ __forceinline__ v16h frag_b(const _Float16* base, int ld, int lane) {
  const int n  = lane & 15;
  const int k0 = (lane >> 4) << 4;
  v16h b;
#pragma unroll
  for (int e = 0; e < 16; ++e) b[e] = base[(k0 + e) * ld + n];
  return b;
}

// B fragment with the matrix stored transposed in LDS: element(k,n)=base[n*ld+k]
__device__ __forceinline__ v16h frag_bT(const _Float16* base, int ld, int lane) {
  const int n  = lane & 15;
  const int k0 = (lane >> 4) << 4;
  v16h b;
#pragma unroll
  for (int e = 0; e < 16; ++e) b[e] = base[n * ld + k0 + e];
  return b;
}

// ---------------------------------------------------------------------------
// Embedding * sqrt(C) + sinusoidal positional encoding
// ---------------------------------------------------------------------------
__global__ __launch_bounds__(256)
void embed_k(const int* __restrict__ tok, const float* __restrict__ emb,
             float* __restrict__ out) {
  const int row = blockIdx.x;          // b*T + t
  const int t   = row & (T - 1);
  const int id  = tok[row];
  const float scale = 22.62741699796952f;       // sqrt(512)
  const float ln1e4 = 9.210340371976184f;       // ln(10000)
#pragma unroll
  for (int j = 0; j < 2; ++j) {
    const int c = threadIdx.x + j * 256;
    float pe;
    if (c < C / 2) {
      const float rate = __expf(-((float)c / (C / 2)) * ln1e4);
      pe = sinf((float)t * rate);
    } else {
      const float rate = __expf(-((float)(c - C / 2) / (C / 2)) * ln1e4);
      pe = cosf((float)t * rate);
    }
    out[(size_t)row * C + c] = emb[(size_t)id * C + c] * scale + pe;
  }
}

// ---------------------------------------------------------------------------
// LayerNorm over C=512, one block (256 thr) per row. gb = [gamma(C), beta(C)]
// ---------------------------------------------------------------------------
__global__ __launch_bounds__(256)
void layernorm_k(const float* __restrict__ x, const float* __restrict__ gb,
                 float* __restrict__ out) {
  __shared__ float s1[8], s2[8];
  const int row = blockIdx.x;
  const int tid = threadIdx.x;
  const float* xr = x + (size_t)row * C;
  float v0 = xr[tid], v1 = xr[tid + 256];
  float s = v0 + v1;
#pragma unroll
  for (int off = 16; off; off >>= 1) s += __shfl_xor(s, off, 32);
  if ((tid & 31) == 0) s1[tid >> 5] = s;
  __syncthreads();
  float mu = 0.f;
#pragma unroll
  for (int i = 0; i < 8; ++i) mu += s1[i];
  mu *= (1.0f / C);
  const float d0 = v0 - mu, d1 = v1 - mu;
  float vs = d0 * d0 + d1 * d1;
#pragma unroll
  for (int off = 16; off; off >>= 1) vs += __shfl_xor(vs, off, 32);
  if ((tid & 31) == 0) s2[tid >> 5] = vs;
  __syncthreads();
  float var = 0.f;
#pragma unroll
  for (int i = 0; i < 8; ++i) var += s2[i];
  var *= (1.0f / C);
  const float rstd = rsqrtf(var + 1e-5f);
  out[(size_t)row * C + tid]       = d0 * rstd * gb[tid]       + gb[C + tid];
  out[(size_t)row * C + tid + 256] = d1 * rstd * gb[tid + 256] + gb[C + tid + 256];
}

// ---------------------------------------------------------------------------
// fp32 -> f16 bulk convert (for TDM-fed weights)
// ---------------------------------------------------------------------------
__global__ __launch_bounds__(256)
void cvt_f16_k(const float* __restrict__ in, _Float16* __restrict__ out) {
  const size_t i = ((size_t)blockIdx.x * 256 + threadIdx.x) * 4;
  *(h4*)&out[i] = cvt4(*(const f4*)&in[i]);
}

// ---------------------------------------------------------------------------
// Generic WMMA GEMM: Out[M, n0..n0+N) = act(A[M,K] @ B + bias) (+ res).
// Block tile 64x128, 4 waves, each wave a 32x64 strip (8 v_wmma per K-chunk).
// QKV mode: B is [3,H,C,D]; column n -> offset (n>>6)*C*D + k*D + (n&63).
// ---------------------------------------------------------------------------
constexpr int LDA_S = 36;    // halfs/row (32 + pad), row = 72B (8B aligned)
constexpr int LDB_S = 136;   // halfs/row (128 + pad), row = 272B

template <bool QKV, bool RELU, bool RES>
__global__ __launch_bounds__(128)
void gemm_wmma(const float* __restrict__ A, const float* __restrict__ Bm,
               const float* __restrict__ bias, const float* __restrict__ res,
               float* __restrict__ Out, int K, int ldb, int ldout, int n0) {
  __shared__ _Float16 As[64 * LDA_S];
  __shared__ _Float16 Bs[32 * LDB_S];
  const int tid  = threadIdx.x;
  const int lane = tid & 31;
  const int wave = tid >> 5;
  const int m0   = blockIdx.y * 64;
  const int nblk = blockIdx.x * 128;
  const int wm   = (wave >> 1) * 32;   // 0 / 32
  const int wn   = (wave & 1) * 64;    // 0 / 64

  v8f acc[2][4] = {};

  for (int k0 = 0; k0 < K; k0 += 32) {
    __syncthreads();
    // stage A 64x32 : 512 float4, 4 per thread
#pragma unroll
    for (int i = 0; i < 4; ++i) {
      const int idx = tid + i * 128;
      const int r = idx >> 3, c = (idx & 7) * 4;
      *(h4*)&As[r * LDA_S + c] = cvt4(*(const f4*)&A[(size_t)(m0 + r) * K + (k0 + c)]);
    }
    // stage B 32x128 : 1024 float4, 8 per thread
#pragma unroll
    for (int i = 0; i < 8; ++i) {
      const int idx = tid + i * 128;
      const int r = idx >> 5, c = (idx & 31) * 4;
      const int ng = n0 + nblk + c;
      const float* src;
      if (QKV)
        src = &Bm[(size_t)(ng >> 6) * (size_t)(C * D) + (size_t)(k0 + r) * D + (ng & 63)];
      else
        src = &Bm[(size_t)(k0 + r) * ldb + ng];
      *(h4*)&Bs[r * LDB_S + c] = cvt4(*(const f4*)src);
    }
    if (k0 + 32 < K)  // speculative prefetch of the next A chunk
      __builtin_prefetch(&A[(size_t)(m0 + (tid >> 1)) * K + (k0 + 32 + (tid & 1) * 16)], 0, 1);
    __syncthreads();

    const v16h a0 = frag_a(&As[wm * LDA_S], LDA_S, lane);
    const v16h a1 = frag_a(&As[(wm + 16) * LDA_S], LDA_S, lane);
    v16h bf[4];
#pragma unroll
    for (int j = 0; j < 4; ++j) bf[j] = frag_b(&Bs[wn + j * 16], LDB_S, lane);
#pragma unroll
    for (int j = 0; j < 4; ++j) {
      acc[0][j] = wmma16(a0, bf[j], acc[0][j]);
      acc[1][j] = wmma16(a1, bf[j], acc[1][j]);
    }
  }

  const int nl = lane & 15;
  const int hb = (lane >> 4) << 3;
#pragma unroll
  for (int ti = 0; ti < 2; ++ti)
#pragma unroll
    for (int tj = 0; tj < 4; ++tj)
#pragma unroll
      for (int r = 0; r < 8; ++r) {
        const int mrow = m0 + wm + ti * 16 + r + hb;
        const int ng   = n0 + nblk + wn + tj * 16 + nl;
        float v = acc[ti][tj][r] + bias[ng];
        if (RELU) v = fmaxf(v, 0.f);
        if (RES) v += res[(size_t)mrow * ldout + ng];
        Out[(size_t)mrow * ldout + ng] = v;
      }
}

// ---------------------------------------------------------------------------
// Vocab-projection GEMM with Tensor-Data-Mover B staging.
// B16 = wf in f16 [K=512][N=32000]. Each K-chunk the 32x128 f16 B tile is
// DMA'd into LDS by TENSOR_LOAD_TO_LDS with LDS padding (16B per 256B row)
// so rows land at the LDB_S=136-half stride the fragment loader expects.
// D# per CDNA5 ISA; issued by wave 0 only (TDM ignores
// EXEC), completion via s_wait_tensorcnt + workgroup barrier.
// 6-arg builtin form (clang-23 / amdgpu-toolchain): the 5th int32x8 carries
// the extended/gather descriptor words (unused here -> zeros).
// ---------------------------------------------------------------------------
__device__ __forceinline__ void tdm_load_b_tile(const _Float16* gsrc,
                                                unsigned ldsOff) {
  const unsigned long long ga = (unsigned long long)(const void*)gsrc;
  v4u g0;
  g0[0] = 1u;                                   // count=1 (valid descriptor)
  g0[1] = ldsOff;                               // lds_addr (bytes)
  g0[2] = (unsigned)ga;                         // global_addr[31:0]
  g0[3] = (unsigned)((ga >> 32) & 0x01FFFFFFu)  // global_addr[56:32]
          | 0x80000000u;                        // type=2 ("image")
  v8i g1;
  g1[0] = (int)0x07510000u;   // data_size=2B | pad_enable | interval=256B | amount=16B
  g1[1] = (int)((unsigned)(V & 0xFFFF) << 16);          // tensor_dim0 lo16
  g1[2] = (int)(((unsigned)V >> 16) | ((unsigned)C << 16)); // dim0 hi | tensor_dim1 lo
  g1[3] = (int)(128u << 16);                            // dim1 hi=0 | tile_dim0=128
  g1[4] = (int)32u;                                     // tile_dim1=32 | tile_dim2=0
  g1[5] = (int)V;                                       // tensor_dim0_stride lo32
  g1[6] = 0;                                            // stride hi | dim1_stride lo
  g1[7] = 0;
  const v4i z4 = {0, 0, 0, 0};
  const v8i z8 = {0, 0, 0, 0, 0, 0, 0, 0};
  __builtin_amdgcn_tensor_load_to_lds(g0, g1, z4, z4, z8, 0);
}

__global__ __launch_bounds__(128)
void gemm_vocab_tdm(const float* __restrict__ A, const _Float16* __restrict__ B16,
                    const float* __restrict__ bias, float* __restrict__ Out) {
  __shared__ _Float16 As[64 * LDA_S];
  __shared__ _Float16 Bs[32 * LDB_S];
  const int tid  = threadIdx.x;
  const int lane = tid & 31;
  const int wave = tid >> 5;
  const int m0   = blockIdx.y * 64;
  const int nblk = blockIdx.x * 128;
  const int wm   = (wave >> 1) * 32;
  const int wn   = (wave & 1) * 64;
  const unsigned bOff = (unsigned)(size_t)(void*)&Bs[0];

  v8f acc[2][4] = {};

  for (int k0 = 0; k0 < C; k0 += 32) {
    __syncthreads();
    // stage A 64x32 (fp32 -> f16, needs conversion so stays on the VALU path)
#pragma unroll
    for (int i = 0; i < 4; ++i) {
      const int idx = tid + i * 128;
      const int r = idx >> 3, c = (idx & 7) * 4;
      *(h4*)&As[r * LDA_S + c] = cvt4(*(const f4*)&A[(size_t)(m0 + r) * C + (k0 + c)]);
    }
    // B tile 32x128 f16 via Tensor Data Mover (wave 0 issues, EXEC ignored)
    if (wave == 0) {
      tdm_load_b_tile(&B16[(size_t)k0 * V + nblk], bOff);
      __builtin_amdgcn_s_wait_tensorcnt(0);
    }
    __syncthreads();

    const v16h a0 = frag_a(&As[wm * LDA_S], LDA_S, lane);
    const v16h a1 = frag_a(&As[(wm + 16) * LDA_S], LDA_S, lane);
    v16h bf[4];
#pragma unroll
    for (int j = 0; j < 4; ++j) bf[j] = frag_b(&Bs[wn + j * 16], LDB_S, lane);
#pragma unroll
    for (int j = 0; j < 4; ++j) {
      acc[0][j] = wmma16(a0, bf[j], acc[0][j]);
      acc[1][j] = wmma16(a1, bf[j], acc[1][j]);
    }
  }

  const int nl = lane & 15;
  const int hb = (lane >> 4) << 3;
#pragma unroll
  for (int ti = 0; ti < 2; ++ti)
#pragma unroll
    for (int tj = 0; tj < 4; ++tj)
#pragma unroll
      for (int r = 0; r < 8; ++r) {
        const int mrow = m0 + wm + ti * 16 + r + hb;
        const int ng   = nblk + wn + tj * 16 + nl;
        Out[(size_t)mrow * V + ng] = acc[ti][tj][r] + bias[ng];
      }
}

// ---------------------------------------------------------------------------
// Flash attention (WMMA QK^T and PV, online softmax).
// ---------------------------------------------------------------------------
constexpr int LDQ = 72, LDK = 72, LDV = 72, LDP = 34;

__global__ __launch_bounds__(128)
void attn_wmma(const float* __restrict__ qbuf, const float* __restrict__ kvbuf,
               float* __restrict__ out, int causal) {
  __shared__ _Float16 Qs[64 * LDQ];
  __shared__ _Float16 Ks[32 * LDK];
  __shared__ _Float16 Vs[32 * LDV];
  __shared__ _Float16 Ps[4][16 * LDP];

  const int tid  = threadIdx.x;
  const int lane = tid & 31;
  const int wave = tid >> 5;
  const int h    = blockIdx.y;
  const int b    = blockIdx.z;
  const int tb0  = blockIdx.x * 64;

  const size_t qbase = (size_t)(b * T + tb0) * NQKV + h * D;
  const size_t kbase = (size_t)(b * T) * NQKV + (size_t)(H * D) + h * D;
  const size_t vbase = (size_t)(b * T) * NQKV + (size_t)(2 * H * D) + h * D;

#pragma unroll
  for (int i = 0; i < 8; ++i) {
    const int idx = tid + i * 128;
    const int r = idx >> 4, c = (idx & 15) * 4;
    *(h4*)&Qs[r * LDQ + c] = cvt4(*(const f4*)&qbuf[qbase + (size_t)r * NQKV + c]);
  }
  __syncthreads();

  const _Float16* qb = &Qs[(wave * 16) * LDQ];
  const v16h qa0 = frag_a(qb, LDQ, lane);
  const v16h qa1 = frag_a(qb + 32, LDQ, lane);

  v8f o[4] = {};
  float Mrun[8], Lrun[8];
#pragma unroll
  for (int r = 0; r < 8; ++r) { Mrun[r] = -1e30f; Lrun[r] = 0.f; }

  const int t0   = tb0 + wave * 16;
  const int sEnd = causal ? (tb0 + 64) : T;   // uniform across the block
  const int nl   = lane & 15;
  const int hb   = (lane >> 4) << 3;

  for (int s0 = 0; s0 < sEnd; s0 += 32) {
    __syncthreads();
#pragma unroll
    for (int i = 0; i < 4; ++i) {
      const int idx = tid + i * 128;
      const int r = idx >> 4, c = (idx & 15) * 4;
      *(h4*)&Ks[r * LDK + c] = cvt4(*(const f4*)&kvbuf[kbase + (size_t)(s0 + r) * NQKV + c]);
      *(h4*)&Vs[r * LDV + c] = cvt4(*(const f4*)&kvbuf[vbase + (size_t)(s0 + r) * NQKV + c]);
    }
    __syncthreads();

    v8f S[2];
#pragma unroll
    for (int jt = 0; jt < 2; ++jt) {
      v8f z = {};
      z = wmma16(qa0, frag_bT(&Ks[(jt * 16) * LDK], LDK, lane), z);
      z = wmma16(qa1, frag_bT(&Ks[(jt * 16) * LDK + 32], LDK, lane), z);
      S[jt] = z;
    }
#pragma unroll
    for (int jt = 0; jt < 2; ++jt)
#pragma unroll
      for (int r = 0; r < 8; ++r) {
        float v = S[jt][r] * 8.0f;   // x sqrt(D), faithful to reference
        if (causal) {
          const int tg = t0 + r + hb;
          const int sg = s0 + jt * 16 + nl;
          if (sg > tg) v = -1e30f;
        }
        S[jt][r] = v;
      }
    float alpha[8];
#pragma unroll
    for (int r = 0; r < 8; ++r) {
      float mx = fmaxf(S[0][r], S[1][r]);
#pragma unroll
      for (int off = 1; off < 16; off <<= 1) mx = fmaxf(mx, __shfl_xor(mx, off, 32));
      const float Mn = fmaxf(Mrun[r], mx);
      const float al = __expf(Mrun[r] - Mn);
      const float p0 = __expf(S[0][r] - Mn);
      const float p1 = __expf(S[1][r] - Mn);
      float sm = p0 + p1;
#pragma unroll
      for (int off = 1; off < 16; off <<= 1) sm += __shfl_xor(sm, off, 32);
      Lrun[r] = Lrun[r] * al + sm;
      Mrun[r] = Mn;
      alpha[r] = al;
      S[0][r] = p0;
      S[1][r] = p1;
    }
    _Float16* pw = Ps[wave];
#pragma unroll
    for (int jt = 0; jt < 2; ++jt)
#pragma unroll
      for (int r = 0; r < 8; ++r)
        pw[(r + hb) * LDP + jt * 16 + nl] = (_Float16)S[jt][r];
#pragma unroll
    for (int nt = 0; nt < 4; ++nt)
#pragma unroll
      for (int r = 0; r < 8; ++r) o[nt][r] *= alpha[r];
    __syncthreads();

    const v16h pa = frag_a(pw, LDP, lane);
#pragma unroll
    for (int nt = 0; nt < 4; ++nt)
      o[nt] = wmma16(pa, frag_b(&Vs[nt * 16], LDV, lane), o[nt]);
  }

#pragma unroll
  for (int nt = 0; nt < 4; ++nt)
#pragma unroll
    for (int r = 0; r < 8; ++r) {
      const int tg = t0 + r + hb;
      const float inv = 1.0f / Lrun[r];
      out[(size_t)(b * T + tg) * C + h * D + nt * 16 + nl] = o[nt][r] * inv;
    }
}

// ---------------------------------------------------------------------------
// Orchestration
// ---------------------------------------------------------------------------
extern "C" void kernel_launch(void* const* d_in, const int* in_sizes, int n_in,
                              void* d_out, int out_size, void* d_ws, size_t ws_size,
                              hipStream_t stream) {
  (void)in_sizes; (void)n_in; (void)out_size; (void)ws_size;

  const int*   src        = (const int*)d_in[0];
  const int*   tgt        = (const int*)d_in[1];
  const float* src_emb    = (const float*)d_in[2];
  const float* tgt_emb    = (const float*)d_in[3];
  const float* enc_wqkv   = (const float*)d_in[4];
  const float* enc_bqkv   = (const float*)d_in[5];
  const float* enc_wo     = (const float*)d_in[6];
  const float* enc_bo     = (const float*)d_in[7];
  const float* enc_ln1    = (const float*)d_in[8];
  const float* enc_ln2    = (const float*)d_in[9];
  const float* enc_ffn_w1 = (const float*)d_in[10];
  const float* enc_ffn_b1 = (const float*)d_in[11];
  const float* enc_ffn_w2 = (const float*)d_in[12];
  const float* enc_ffn_b2 = (const float*)d_in[13];
  const float* dec_swqkv  = (const float*)d_in[14];
  const float* dec_sbqkv  = (const float*)d_in[15];
  const float* dec_w1     = (const float*)d_in[16];
  const float* dec_b1     = (const float*)d_in[17];
  const float* dec_ln1    = (const float*)d_in[18];
  const float* dec_cwqkv  = (const float*)d_in[19];
  const float* dec_cbqkv  = (const float*)d_in[20];
  const float* dec_w2     = (const float*)d_in[21];
  const float* dec_b2     = (const float*)d_in[22];
  const float* dec_ln2    = (const float*)d_in[23];
  const float* dec_ln3    = (const float*)d_in[24];
  const float* dec_ffn_w1 = (const float*)d_in[25];
  const float* dec_ffn_b1 = (const float*)d_in[26];
  const float* dec_ffn_w2 = (const float*)d_in[27];
  const float* dec_ffn_b2 = (const float*)d_in[28];
  const float* wf         = (const float*)d_in[29];
  const float* bf         = (const float*)d_in[30];

  float* p    = (float*)d_ws;
  float* x    = p; p += (size_t)M * C;      // encoder state -> enc_out
  float* y    = p; p += (size_t)M * C;      // decoder state
  float* xn   = p; p += (size_t)M * C;      // normed activations
  float* att  = p; p += (size_t)M * C;      // attention output
  float* hid  = p; p += (size_t)M * F;      // FFN hidden
  float* qkv1 = p; p += (size_t)M * NQKV;   // fused qkv (q side)
  float* qkv2 = p; p += (size_t)M * NQKV;   // fused qkv (kv side, cross)
  _Float16* wf16 = (_Float16*)p;            // f16 copy of wf [C, V] for TDM

  const dim3 blk(128);
  const dim3 gAttn(T / 64, H, B);
  auto ggrid = [](int n) { return dim3(n / 128, M / 64); };

  // f16 weight copy for the TDM-fed vocab projection (C*V / 1024 blocks)
  cvt_f16_k<<<(C * V) / 1024, 256, 0, stream>>>(wf, wf16);

  // ---------------- encoder ----------------
  embed_k<<<M, 256, 0, stream>>>(src, src_emb, x);
  for (int l = 0; l < L; ++l) {
    const float* wqkv = enc_wqkv + (size_t)l * 3 * H * C * D;
    const float* bqkv = enc_bqkv + (size_t)l * 3 * H * D;
    layernorm_k<<<M, 256, 0, stream>>>(x, enc_ln1 + (size_t)l * 2 * C, xn);
    gemm_wmma<true, false, false><<<ggrid(NQKV), blk, 0, stream>>>(
        xn, wqkv, bqkv, nullptr, qkv1, C, 0, NQKV, 0);
    attn_wmma<<<gAttn, blk, 0, stream>>>(qkv1, qkv1, att, 0);
    gemm_wmma<false, false, true><<<ggrid(C), blk, 0, stream>>>(
        att, enc_wo + (size_t)l * C * C, enc_bo + (size_t)l * C, x, x, C, C, C, 0);
    layernorm_k<<<M, 256, 0, stream>>>(x, enc_ln2 + (size_t)l * 2 * C, xn);
    gemm_wmma<false, true, false><<<ggrid(F), blk, 0, stream>>>(
        xn, enc_ffn_w1 + (size_t)l * C * F, enc_ffn_b1 + (size_t)l * F,
        nullptr, hid, C, F, F, 0);
    gemm_wmma<false, false, true><<<ggrid(C), blk, 0, stream>>>(
        hid, enc_ffn_w2 + (size_t)l * F * C, enc_ffn_b2 + (size_t)l * C,
        x, x, F, C, C, 0);
  }
  // x == enc_out from here on

  // ---------------- decoder ----------------
  embed_k<<<M, 256, 0, stream>>>(tgt, tgt_emb, y);
  for (int l = 0; l < L; ++l) {
    const float* swqkv = dec_swqkv + (size_t)l * 3 * H * C * D;
    const float* sbqkv = dec_sbqkv + (size_t)l * 3 * H * D;
    const float* cwqkv = dec_cwqkv + (size_t)l * 3 * H * C * D;
    const float* cbqkv = dec_cbqkv + (size_t)l * 3 * H * D;

    layernorm_k<<<M, 256, 0, stream>>>(y, dec_ln1 + (size_t)l * 2 * C, xn);
    gemm_wmma<true, false, false><<<ggrid(NQKV), blk, 0, stream>>>(
        xn, swqkv, sbqkv, nullptr, qkv1, C, 0, NQKV, 0);
    attn_wmma<<<gAttn, blk, 0, stream>>>(qkv1, qkv1, att, 1);
    gemm_wmma<false, false, true><<<ggrid(C), blk, 0, stream>>>(
        att, dec_w1 + (size_t)l * C * C, dec_b1 + (size_t)l * C, y, y, C, C, C, 0);

    layernorm_k<<<M, 256, 0, stream>>>(y, dec_ln2 + (size_t)l * 2 * C, xn);
    gemm_wmma<true, false, false><<<dim3((H * D) / 128, M / 64), blk, 0, stream>>>(
        xn, cwqkv, cbqkv, nullptr, qkv1, C, 0, NQKV, 0);           // q cols [0,512)
    gemm_wmma<true, false, false><<<dim3((2 * H * D) / 128, M / 64), blk, 0, stream>>>(
        x, cwqkv, cbqkv, nullptr, qkv2, C, 0, NQKV, H * D);        // k,v cols [512,1536)
    attn_wmma<<<gAttn, blk, 0, stream>>>(qkv1, qkv2, att, 0);
    gemm_wmma<false, false, true><<<ggrid(C), blk, 0, stream>>>(
        att, dec_w2 + (size_t)l * C * C, dec_b2 + (size_t)l * C, y, y, C, C, C, 0);

    layernorm_k<<<M, 256, 0, stream>>>(y, dec_ln3 + (size_t)l * 2 * C, xn);
    gemm_wmma<false, true, false><<<ggrid(F), blk, 0, stream>>>(
        xn, dec_ffn_w1 + (size_t)l * C * F, dec_ffn_b1 + (size_t)l * F,
        nullptr, hid, C, F, F, 0);
    gemm_wmma<false, false, true><<<ggrid(C), blk, 0, stream>>>(
        hid, dec_ffn_w2 + (size_t)l * F * C, dec_ffn_b2 + (size_t)l * C,
        y, y, F, C, C, 0);
  }

  // ---------------- vocab projection (TDM-staged B) ----------------
  gemm_vocab_tdm<<<dim3(V / 128, M / 64), blk, 0, stream>>>(
      y, wf16, bf, (float*)d_out);
}
